// AxialAttentionLayer_54546084659577
// MI455X (gfx1250) — matrix operand order repaired and checked by reading, not compiled, not run on hardware
//
#include <hip/hip_runtime.h>
#include <hip/hip_bf16.h>

// ---------------------------------------------------------------------------
// AxialAttentionLayer for MI455X (gfx1250, wave32, WMMA bf16 path)
// LDS tiles stored pre-converted to bf16 in WMMA-K-packed layout so each
// operand VGPR is one 32-bit ds_load (no per-gather VALU conversion).
// ---------------------------------------------------------------------------

typedef __bf16 v16bf __attribute__((ext_vector_type(16)));
typedef float  v8f   __attribute__((ext_vector_type(8)));

union V16U { v16bf bf; unsigned u[8]; };

#define B_      8
#define CIN_    256
#define O2_     512        // 2*COUT
#define SPAN_   64
#define EPS_    1e-5f
#define NBB     512        // B*H
#define QKV_ELEMS (NBB * O2_ * SPAN_)   // 16,777,216 floats

// stats workspace layout (floats)
#define ST_QKV_SCALE 0
#define ST_QKV_SHIFT 512
#define ST_SIM_SUM   1024   // 48
#define ST_SIM_SQ    1072   // 48
#define ST_SIM_SCALE 1120   // 48
#define ST_SIM_SHIFT 1168   // 48
#define ST_OUT_SUM   1216   // 512
#define ST_OUT_SQ    1728   // 512
#define ST_OUT_SCALE 2240   // 512
#define ST_OUT_SHIFT 2752   // 512
#define ST_TOTAL     3264

// ---- bf16 helpers ----------------------------------------------------------
__device__ __forceinline__ unsigned short f2bfu(float f) {
  __bf16 h = (__bf16)f;                       // native RNE convert
  unsigned short u; __builtin_memcpy(&u, &h, 2); return u;
}
__device__ __forceinline__ unsigned f2bf_pk(float a, float b) {
  return (unsigned)f2bfu(a) | ((unsigned)f2bfu(b) << 16);
}
__device__ __forceinline__ float bf2f(unsigned short h) {
  union { unsigned u; float f; } x; x.u = ((unsigned)h) << 16;
  return x.f;
}

// A-operand VGPR v holds K pair base: v<4 -> 2v (+8 hi), v>=4 -> 16+2(v-4) (+8 hi)
__device__ __forceinline__ int kAp(int v, bool hi) {
  return ((v >> 2) << 4) + ((v & 3) << 1) + (hi ? 8 : 0);
}
// B-operand VGPR v holds K pair base: v<4 -> 2v (+8 hi), v>=4 -> 16+2(v-4) (+8 hi)
__device__ __forceinline__ int kBp(int v, bool hi) {
  return ((v >> 2) << 4) + ((v & 3) << 1) + (hi ? 8 : 0);
}

__device__ __forceinline__ float block_reduce(float v, float* red) {
  int t = threadIdx.x;
  red[t] = v; __syncthreads();
  for (int s = 128; s > 0; s >>= 1) {
    if (t < s) red[t] += red[t + s];
    __syncthreads();
  }
  float r = red[0]; __syncthreads();
  return r;
}

__device__ __forceinline__ v8f wmma_bf16(const v16bf& a, const v16bf& b, v8f c) {
  return __builtin_amdgcn_wmma_f32_16x16x32_bf16(
      false, a, false, b, (short)0, c, false, false);
}

// ---------------------------------------------------------------------------
// K0: zero the stats region
// ---------------------------------------------------------------------------
__global__ void k_zero(float* __restrict__ p, int n) {
  int i = blockIdx.x * 256 + threadIdx.x;
  if (i < n) p[i] = 0.f;
}

// ---------------------------------------------------------------------------
// K1: qkv[bb,o,w] = sum_c W[o,c]*xi[bb,c,w]  (512 batched GEMMs via WMMA)
// ---------------------------------------------------------------------------
__global__ __launch_bounds__(256)
void k_qkv_gemm(const float* __restrict__ x, const float* __restrict__ Wq,
                float* __restrict__ qkv) {
  __shared__ unsigned short sWb[128 * 32];   // [m][k] bf16, 8 KB
  __shared__ unsigned short sXb[64 * 32];    // [n][k] bf16 (transposed), 4 KB
  const int tid = threadIdx.x;
  const int wave = tid >> 5, lane = tid & 31, lm = lane & 15;
  const bool hi = lane >= 16;
  const int bb = blockIdx.y;
  const int Mb = blockIdx.x * 128;
  const long xbase = (long)(bb >> 6) * (CIN_ * 4096) + (long)(bb & 63) * 64;
  v8f acc[4] = {};
  for (int k0 = 0; k0 < CIN_; k0 += 32) {
#pragma unroll
    for (int q = 0; q < 4; ++q) {            // stage W tile 128x32 (b128 loads)
      int flat = (tid + 256 * q) * 4;
      int r = flat >> 5, c = flat & 31;
      float4 v = *reinterpret_cast<const float4*>(&Wq[(Mb + r) * CIN_ + k0 + c]);
      *reinterpret_cast<unsigned*>(&sWb[r * 32 + c])     = f2bf_pk(v.x, v.y);
      *reinterpret_cast<unsigned*>(&sWb[r * 32 + c + 2]) = f2bf_pk(v.z, v.w);
    }
#pragma unroll
    for (int q = 0; q < 2; ++q) {            // stage X tile 32x64 -> [n][k]
      int flat = (tid + 256 * q) * 4;
      int r = flat >> 6, c = flat & 63;
      float4 v = *reinterpret_cast<const float4*>(&x[xbase + (long)(k0 + r) * 4096 + c]);
      sXb[(c + 0) * 32 + r] = f2bfu(v.x);
      sXb[(c + 1) * 32 + r] = f2bfu(v.y);
      sXb[(c + 2) * 32 + r] = f2bfu(v.z);
      sXb[(c + 3) * 32 + r] = f2bfu(v.w);
    }
    __syncthreads();
    V16U a;
#pragma unroll
    for (int v = 0; v < 8; ++v)
      a.u[v] = *reinterpret_cast<const unsigned*>(&sWb[(wave * 16 + lm) * 32 + kAp(v, hi)]);
#pragma unroll
    for (int nt = 0; nt < 4; ++nt) {
      V16U b;
#pragma unroll
      for (int v = 0; v < 8; ++v)
        b.u[v] = *reinterpret_cast<const unsigned*>(&sXb[(nt * 16 + lm) * 32 + kBp(v, hi)]);
      acc[nt] = wmma_bf16(a.bf, b.bf, acc[nt]);
    }
    __syncthreads();
  }
#pragma unroll
  for (int nt = 0; nt < 4; ++nt)
#pragma unroll
    for (int r = 0; r < 8; ++r) {
      int o = Mb + wave * 16 + (hi ? 8 : 0) + r;
      int w = nt * 16 + lm;
      qkv[(long)bb * 32768 + o * 64 + w] = acc[nt][r];
    }
}

// ---------------------------------------------------------------------------
// K2: per-channel BN stats for qkv
// ---------------------------------------------------------------------------
__global__ __launch_bounds__(256)
void k_qkv_stats(const float* __restrict__ qkv, const float* __restrict__ g,
                 const float* __restrict__ bta, float* __restrict__ stat) {
  __shared__ float red[256];
  const int o = blockIdx.x;
  const int tid = threadIdx.x;
  float s = 0.f, q = 0.f;
  for (int t = 0; t < 128; ++t) {
    int j = tid + 256 * t;
    float v = qkv[(long)(j >> 6) * 32768 + o * 64 + (j & 63)];
    s += v; q += v * v;
  }
  s = block_reduce(s, red);
  q = block_reduce(q, red);
  if (tid == 0) {
    const float inv = 1.f / 32768.f;
    float mean = s * inv;
    float var  = q * inv - mean * mean;
    float sc   = rsqrtf(var + EPS_) * g[o];
    stat[ST_QKV_SCALE + o] = sc;
    stat[ST_QKV_SHIFT + o] = bta[o] - mean * sc;
  }
}

// ---------------------------------------------------------------------------
// K3: apply qkv BN in place
// ---------------------------------------------------------------------------
__global__ __launch_bounds__(256)
void k_qkv_apply(float* __restrict__ qkv, const float* __restrict__ stat) {
  long i = (long)blockIdx.x * 256 + threadIdx.x;
  int o = (int)((i >> 6) & 511);
  qkv[i] = qkv[i] * stat[ST_QKV_SCALE + o] + stat[ST_QKV_SHIFT + o];
}

// ---------------------------------------------------------------------------
// K4: fused stats pass per (g, bb): qe/ke (VALU) + qk (WMMA), stats only
// ---------------------------------------------------------------------------
__global__ __launch_bounds__(256)
void k_sim_stats(const float* __restrict__ qkv, const float* __restrict__ rel,
                 float* __restrict__ stat) {
  __shared__ float sEmb[16 * 127];
  __shared__ float sQK[16 * 64];              // q(8ch)+k(8ch) fp32
  __shared__ unsigned short sQeT[64 * 64];    // qe^T: [j][t] bf16, K-packed
  __shared__ unsigned short sKeT[64 * 64];    // ke^T: [j][t] bf16, K-packed
  __shared__ float red[256];
  const int tid = threadIdx.x;
  const int g = blockIdx.x, bb = blockIdx.y;
  const int wave = tid >> 5, lane = tid & 31, lm = lane & 15;
  const bool hi = lane >= 16;
  for (int t = 0; t < 8; ++t) { int i = tid + 256 * t; if (i < 16 * 127) sEmb[i] = rel[i]; }
  {
    int flat = tid * 4;                        // 1024 floats, float4 loads
    int c = flat >> 6, w = flat & 63;
    float4 v = *reinterpret_cast<const float4*>(
        &qkv[(long)bb * 32768 + (g * 32 + c) * 64 + w]);
    *reinterpret_cast<float4*>(&sQK[flat]) = v;
  }
  __syncthreads();
  float qe_s = 0, qe_q = 0, ke_s = 0, ke_q = 0;
  {
    int i = tid >> 2, j0 = (tid & 3) << 4;     // i = qe row t
    for (int jj = 0; jj < 16; ++jj) {
      int j = j0 + jj, d = i - j + 63;
      float aq = 0.f, ak = 0.f;
#pragma unroll
      for (int c = 0; c < 8; ++c) {
        aq += sQK[c * 64 + i]       * sEmb[c * 127 + d];
        ak += sQK[(8 + c) * 64 + i] * sEmb[(8 + c) * 127 + d];
      }
      sQeT[j * 64 + i] = f2bfu(aq);            // transposed store
      sKeT[j * 64 + i] = f2bfu(ak);
      qe_s += aq; qe_q += aq * aq; ke_s += ak; ke_q += ak * ak;
    }
  }
  __syncthreads();
  float qk_s = 0, qk_q = 0;
  for (int t = 0; t < 2; ++t) {
    int id = wave * 2 + t, ti = id >> 2, tj = id & 3;
    v8f acc = {};
    for (int k0 = 0; k0 < 64; k0 += 32) {
      V16U a, b;
#pragma unroll
      for (int v = 0; v < 8; ++v) {
        a.u[v] = *reinterpret_cast<const unsigned*>(&sQeT[(ti * 16 + lm) * 64 + k0 + kAp(v, hi)]);
        b.u[v] = *reinterpret_cast<const unsigned*>(&sKeT[(tj * 16 + lm) * 64 + k0 + kBp(v, hi)]);
      }
      acc = wmma_bf16(a.bf, b.bf, acc);
    }
#pragma unroll
    for (int r = 0; r < 8; ++r) { float v = acc[r]; qk_s += v; qk_q += v * v; }
  }
  float v;
  v = block_reduce(qk_s, red); if (tid == 0) atomicAdd(&stat[ST_SIM_SUM + g], v);
  v = block_reduce(qk_q, red); if (tid == 0) atomicAdd(&stat[ST_SIM_SQ + g], v);
  v = block_reduce(qe_s, red); if (tid == 0) atomicAdd(&stat[ST_SIM_SUM + 16 + g], v);
  v = block_reduce(qe_q, red); if (tid == 0) atomicAdd(&stat[ST_SIM_SQ + 16 + g], v);
  v = block_reduce(ke_s, red); if (tid == 0) atomicAdd(&stat[ST_SIM_SUM + 32 + g], v);
  v = block_reduce(ke_q, red); if (tid == 0) atomicAdd(&stat[ST_SIM_SQ + 32 + g], v);
}

// ---------------------------------------------------------------------------
// K5: finalize sim BN scale/shift (48 channels)
// ---------------------------------------------------------------------------
__global__ void k_sim_fin(const float* __restrict__ g, const float* __restrict__ b,
                          float* __restrict__ stat) {
  int ch = threadIdx.x;
  if (ch < 48) {
    const float inv = 1.f / (float)(NBB * SPAN_ * SPAN_);
    float mean = stat[ST_SIM_SUM + ch] * inv;
    float var  = stat[ST_SIM_SQ + ch] * inv - mean * mean;
    float sc   = rsqrtf(var + EPS_) * g[ch];
    stat[ST_SIM_SCALE + ch] = sc;
    stat[ST_SIM_SHIFT + ch] = b[ch] - mean * sc;
  }
}

// ---------------------------------------------------------------------------
// K6: fused attention pass per (g, bb)
// ---------------------------------------------------------------------------
__global__ __launch_bounds__(256)
void k_attn(const float* __restrict__ qkv, const float* __restrict__ rel,
            float* __restrict__ stat, float* __restrict__ outp) {
  __shared__ float sEmb[32 * 127];            // 16256 B
  __shared__ float sQK[16 * 64];              // q,k fp32 (4 KB)
  __shared__ unsigned short sVb[16 * 64];     // v bf16 [c][j] (2 KB)
  __shared__ unsigned short sQeT[64 * 64];    // 8 KB
  __shared__ unsigned short sKeT[64 * 64];    // 8 KB
  __shared__ float sSim[64 * 64];             // 16 KB (for attn_e)
  __shared__ unsigned short sSimB[64 * 64];   // 8 KB (bf16 for WMMA)
  __shared__ float sOsum[32];
  __shared__ float sOsq[32];
  const int tid = threadIdx.x;
  const int g = blockIdx.x, bb = blockIdx.y;
  const int wave = tid >> 5, lane = tid & 31, lm = lane & 15;
  const bool hi = lane >= 16;
  if (tid < 32) { sOsum[tid] = 0.f; sOsq[tid] = 0.f; }
  for (int t = 0; t < 16; ++t) { int i = tid + 256 * t; if (i < 32 * 127) sEmb[i] = rel[i]; }
#pragma unroll
  for (int q = 0; q < 2; ++q) {                // 2048 floats, float4 loads
    int flat = (tid + 256 * q) * 4;
    int c = flat >> 6, w = flat & 63;
    float4 v = *reinterpret_cast<const float4*>(
        &qkv[(long)bb * 32768 + (g * 32 + c) * 64 + w]);
    if (c < 16) {
      *reinterpret_cast<float4*>(&sQK[flat]) = v;
    } else {
      int base = (c - 16) * 64 + w;
      *reinterpret_cast<unsigned*>(&sVb[base])     = f2bf_pk(v.x, v.y);
      *reinterpret_cast<unsigned*>(&sVb[base + 2]) = f2bf_pk(v.z, v.w);
    }
  }
  __syncthreads();
  {
    int i = tid >> 2, j0 = (tid & 3) << 4;
    for (int jj = 0; jj < 16; ++jj) {
      int j = j0 + jj, d = i - j + 63;
      float aq = 0.f, ak = 0.f;
#pragma unroll
      for (int c = 0; c < 8; ++c) {
        aq += sQK[c * 64 + i]       * sEmb[c * 127 + d];
        ak += sQK[(8 + c) * 64 + i] * sEmb[(8 + c) * 127 + d];
      }
      sQeT[j * 64 + i] = f2bfu(aq);
      sKeT[j * 64 + i] = f2bfu(ak);
    }
  }
  __syncthreads();
  const float sc_qk = stat[ST_SIM_SCALE + g],      sh_qk = stat[ST_SIM_SHIFT + g];
  const float sc_qe = stat[ST_SIM_SCALE + 16 + g], sh_qe = stat[ST_SIM_SHIFT + 16 + g];
  const float sc_ke = stat[ST_SIM_SCALE + 32 + g], sh_ke = stat[ST_SIM_SHIFT + 32 + g];
  for (int t = 0; t < 2; ++t) {
    int id = wave * 2 + t, ti = id >> 2, tj = id & 3;
    v8f acc = {};
    for (int k0 = 0; k0 < 64; k0 += 32) {
      V16U a, b;
#pragma unroll
      for (int v = 0; v < 8; ++v) {
        a.u[v] = *reinterpret_cast<const unsigned*>(&sQeT[(ti * 16 + lm) * 64 + k0 + kAp(v, hi)]);
        b.u[v] = *reinterpret_cast<const unsigned*>(&sKeT[(tj * 16 + lm) * 64 + k0 + kBp(v, hi)]);
      }
      acc = wmma_bf16(a.bf, b.bf, acc);
    }
#pragma unroll
    for (int r = 0; r < 8; ++r) {
      int I = ti * 16 + (hi ? 8 : 0) + r;
      int J = tj * 16 + lm;
      float qe = bf2f(sQeT[J * 64 + I]);      // qe at its own (i=I, j=J)
      float ke = bf2f(sKeT[J * 64 + I]);
      sSim[I * 64 + J] = acc[r] * sc_qk + sh_qk + qe * sc_qe + sh_qe + ke * sc_ke + sh_ke;
    }
  }
  __syncthreads();
  // softmax over j: wave handles 8 rows, 32 lanes x 2 cols each
  for (int rr = 0; rr < 8; ++rr) {
    int i = wave * 8 + rr;
    float v0 = sSim[i * 64 + lane], v1 = sSim[i * 64 + 32 + lane];
    float m = fmaxf(v0, v1);
#pragma unroll
    for (int s = 16; s > 0; s >>= 1) m = fmaxf(m, __shfl_xor(m, s, 32));
    float e0 = __expf(v0 - m), e1 = __expf(v1 - m);
    float su = e0 + e1;
#pragma unroll
    for (int s = 16; s > 0; s >>= 1) su += __shfl_xor(su, s, 32);
    float inv = 1.f / su;
    float r0 = e0 * inv, r1 = e1 * inv;
    sSim[i * 64 + lane]       = r0;
    sSim[i * 64 + 32 + lane]  = r1;
    sSimB[i * 64 + lane]      = f2bfu(r0);
    sSimB[i * 64 + 32 + lane] = f2bfu(r1);
  }
  __syncthreads();
  const long obase = (long)bb * 32768;
  if (wave < 4) {
    // attn[i,c] = sum_j sim[i,j]*v[c,j] : M=64(i) x N=16(c), K=64(j)
    int ti = wave;
    v8f acc = {};
    for (int k0 = 0; k0 < 64; k0 += 32) {
      V16U a, b;
#pragma unroll
      for (int v = 0; v < 8; ++v) {
        a.u[v] = *reinterpret_cast<const unsigned*>(&sSimB[(ti * 16 + lm) * 64 + k0 + kAp(v, hi)]);
        b.u[v] = *reinterpret_cast<const unsigned*>(&sVb[lm * 64 + k0 + kBp(v, hi)]);
      }
      acc = wmma_bf16(a.bf, b.bf, acc);
    }
#pragma unroll
    for (int r = 0; r < 8; ++r) {
      int i = ti * 16 + (hi ? 8 : 0) + r;
      int c = lm;
      float v = acc[r];
      outp[obase + (g * 32 + 2 * c) * 64 + i] = v;
      atomicAdd(&sOsum[2 * c], v);
      atomicAdd(&sOsq[2 * c], v * v);
    }
  } else {
    // attn_e[c,i] = sum_j sim[i,j]*v_emb[c,i,j]
    int t2 = tid - 128;
    for (int t = 0; t < 8; ++t) {
      int idx = t2 + 128 * t;
      int c = idx >> 6, i = idx & 63;
      float v = 0.f;
      for (int j = 0; j < 64; ++j)
        v += sSim[i * 64 + j] * sEmb[(16 + c) * 127 + i - j + 63];
      outp[obase + (g * 32 + 2 * c + 1) * 64 + i] = v;
      atomicAdd(&sOsum[2 * c + 1], v);
      atomicAdd(&sOsq[2 * c + 1], v * v);
    }
  }
  __syncthreads();
  if (tid < 32) {
    int o = 32 * g + tid;
    atomicAdd(&stat[ST_OUT_SUM + o], sOsum[tid]);
    atomicAdd(&stat[ST_OUT_SQ + o], sOsq[tid]);
  }
}

// ---------------------------------------------------------------------------
// K7: finalize out BN scale/shift (512 channels)
// ---------------------------------------------------------------------------
__global__ void k_out_fin(const float* __restrict__ g, const float* __restrict__ b,
                          float* __restrict__ stat) {
  int o = threadIdx.x;
  const float inv = 1.f / 32768.f;
  float mean = stat[ST_OUT_SUM + o] * inv;
  float var  = stat[ST_OUT_SQ + o] * inv - mean * mean;
  float sc   = rsqrtf(var + EPS_) * g[o];
  stat[ST_OUT_SCALE + o] = sc;
  stat[ST_OUT_SHIFT + o] = b[o] - mean * sc;
}

// ---------------------------------------------------------------------------
// K8: final BN + pair-sum + transpose to (B, COUT, H, W)
// ---------------------------------------------------------------------------
__global__ __launch_bounds__(256)
void k_out_write(const float* __restrict__ outp, const float* __restrict__ stat,
                 float* __restrict__ out) {
  long i = (long)blockIdx.x * 256 + threadIdx.x;
  int w    = (int)(i & 63);
  int h    = (int)((i >> 6) & 63);
  int cout = (int)((i >> 12) & 255);
  int b0   = (int)(i >> 20);
  long bb  = (long)b0 * 64 + h;
  int o0 = cout * 2, o1 = cout * 2 + 1;
  float v0 = outp[bb * 32768 + o0 * 64 + w];
  float v1 = outp[bb * 32768 + o1 * 64 + w];
  out[i] = v0 * stat[ST_OUT_SCALE + o0] + stat[ST_OUT_SHIFT + o0]
         + v1 * stat[ST_OUT_SCALE + o1] + stat[ST_OUT_SHIFT + o1];
}

// ---------------------------------------------------------------------------
extern "C" void kernel_launch(void* const* d_in, const int* in_sizes, int n_in,
                              void* d_out, int out_size, void* d_ws, size_t ws_size,
                              hipStream_t stream) {
  (void)in_sizes; (void)n_in; (void)out_size; (void)ws_size;
  const float* x   = (const float*)d_in[0];
  const float* qw  = (const float*)d_in[1];
  const float* rel = (const float*)d_in[2];
  const float* gq  = (const float*)d_in[3];
  const float* bq  = (const float*)d_in[4];
  const float* gs  = (const float*)d_in[5];
  const float* bs  = (const float*)d_in[6];
  const float* go  = (const float*)d_in[7];
  const float* bo  = (const float*)d_in[8];
  float* out  = (float*)d_out;
  float* qkv  = (float*)d_ws;                 // 64 MB
  float* outp = qkv + (size_t)QKV_ELEMS;      // 64 MB
  float* stat = outp + (size_t)QKV_ELEMS;     // ~13 KB

  k_zero<<<(ST_TOTAL + 255) / 256, 256, 0, stream>>>(stat, ST_TOTAL);
  k_qkv_gemm<<<dim3(4, 512), 256, 0, stream>>>(x, qw, qkv);
  k_qkv_stats<<<512, 256, 0, stream>>>(qkv, gq, bq, stat);
  k_qkv_apply<<<QKV_ELEMS / 256, 256, 0, stream>>>(qkv, stat);
  k_sim_stats<<<dim3(16, 512), 256, 0, stream>>>(qkv, rel, stat);
  k_sim_fin<<<1, 64, 0, stream>>>(gs, bs, stat);
  k_attn<<<dim3(16, 512), 256, 0, stream>>>(qkv, rel, stat, outp);
  k_out_fin<<<1, 512, 0, stream>>>(go, bo, stat);
  k_out_write<<<(B_ * 256 * 64 * 64) / 256, 256, 0, stream>>>(outp, stat, out);
}
